// NetGraphConvolution_33114197852789
// MI455X (gfx1250) — matrix operand order repaired and checked by reading, not compiled, not run on hardware
//
#include <hip/hip_runtime.h>
#include <math.h>

typedef float        v2f   __attribute__((ext_vector_type(2)));
typedef float        v4f   __attribute__((ext_vector_type(4)));
typedef float        v8f   __attribute__((ext_vector_type(8)));
typedef __bf16       v16bf __attribute__((ext_vector_type(16)));
typedef unsigned int v2u   __attribute__((ext_vector_type(2)));
typedef unsigned int v4u   __attribute__((ext_vector_type(4)));
typedef unsigned int v8u   __attribute__((ext_vector_type(8)));

#define NN     8192
#define IN_F   512
#define OUT_F  256
#define KBLK   64
#define MT     2            // 16-row M-tiles per wave
#define MROWS  (16 * MT)    // 32 rows per workgroup
#define PAD32  68           // fp32 LDS row stride: 272B = 17*16, b128-aligned
#define PADH   72           // bf16 LDS row stride: 144B, b128-aligned

// ---- fp32 <-> bf16 split helpers (RNE) ------------------------------------
__device__ __forceinline__ unsigned short f32_to_bf16_rne(float f) {
    unsigned int u = __float_as_uint(f);
    u += 0x7FFFu + ((u >> 16) & 1u);
    return (unsigned short)(u >> 16);
}
__device__ __forceinline__ float bf16_as_f32(unsigned short h) {
    return __uint_as_float((unsigned int)h << 16);
}
__device__ __forceinline__ void split_bf16(float v, unsigned short& hi, unsigned short& lo) {
    hi = f32_to_bf16_rne(v);
    lo = f32_to_bf16_rne(v - bf16_as_f32(hi));
}
__device__ __forceinline__ v16bf make_bf16x16(v4u a, v4u b) {
    v8u w = {a.x, a.y, a.z, a.w, b.x, b.y, b.z, b.w};
    return __builtin_bit_cast(v16bf, w);
}
// split a float4 into packed-bf16 hi/lo dword pairs
__device__ __forceinline__ void split4(v4f v, v2u& h, v2u& l) {
    unsigned short h0, l0, h1, l1, h2, l2, h3, l3;
    split_bf16(v.x, h0, l0); split_bf16(v.y, h1, l1);
    split_bf16(v.z, h2, l2); split_bf16(v.w, h3, l3);
    h.x = (unsigned int)h0 | ((unsigned int)h1 << 16);
    h.y = (unsigned int)h2 | ((unsigned int)h3 << 16);
    l.x = (unsigned int)l0 | ((unsigned int)l1 << 16);
    l.y = (unsigned int)l2 | ((unsigned int)l3 << 16);
}

// ---------------------------------------------------------------------------
// Stage 1: support = X @ W  (exact fp32 WMMA 16x16x4), epilogue splits the
// result into two TRANSPOSED bf16 planes BhT/BlT [OUT_F][NN] for stage 2.
// ---------------------------------------------------------------------------
__global__ __launch_bounds__(512)
void gcn_support_kernel(const float* __restrict__ X,
                        const float* __restrict__ W,
                        unsigned short* __restrict__ BhT,
                        unsigned short* __restrict__ BlT) {
    __shared__ __align__(16) float aT[2][MROWS * PAD32];

    const int tid  = threadIdx.x;
    const int wave = tid >> 5, lane = tid & 31;
    const int laneM = lane & 15;
    const int kOff  = (lane >> 4) << 1;
    const int row0  = blockIdx.x * MROWS;
    const int n0    = wave * 16;
    // stager: one b128 per thread per block: row rs, cols cs..cs+3
    const int rs = tid >> 4;             // 0..31
    const int cs = (tid & 15) * 4;       // 0..60
    const float* aBase  = X + (size_t)(row0 + rs) * IN_F + cs;
    const int    ldsIdx = rs * PAD32 + cs;

    *(v4f*)&aT[0][ldsIdx] = *(const v4f*)aBase;
    __syncthreads();

    v8f acc[MT] = {};
    constexpr int NB = IN_F / KBLK;

    for (int kb = 0; kb < NB; ++kb) {
        const int  cur  = kb & 1;
        const bool more = (kb + 1 < NB);
        v4f stage;
        if (more) stage = *(const v4f*)(aBase + (size_t)(kb + 1) * KBLK);

        const float* at = aT[cur];
        const float* bp = W + (size_t)(kb * KBLK + kOff) * OUT_F + n0 + laneM;

        #pragma unroll
        for (int k = 0; k < KBLK; k += 4) {
            v2f b; b.x = bp[(size_t)k * OUT_F]; b.y = bp[(size_t)(k + 1) * OUT_F];
            #pragma unroll
            for (int t = 0; t < MT; ++t) {
                const float* ap = &at[(16 * t + laneM) * PAD32 + k + kOff];
                v2f a; a.x = ap[0]; a.y = ap[1];
                acc[t] = __builtin_amdgcn_wmma_f32_16x16x4_f32(
                    false, a, false, b, (short)0, acc[t], false, false);
            }
        }
        if (more) *(v4f*)&aT[cur ^ 1][ldsIdx] = stage;
        __syncthreads();
    }

    // epilogue: acc[t][r] = support[row0+16t+mHalf+r][n] -> BhT/BlT[n][m]
    const int mHalf = (lane >> 4) * 8;
    const int n     = n0 + laneM;
    #pragma unroll
    for (int t = 0; t < MT; ++t) {
        unsigned int hw[4], lw[4];
        #pragma unroll
        for (int i = 0; i < 4; ++i) {
            unsigned short h0, l0, h1, l1;
            split_bf16(acc[t][2 * i],     h0, l0);
            split_bf16(acc[t][2 * i + 1], h1, l1);
            hw[i] = (unsigned int)h0 | ((unsigned int)h1 << 16);
            lw[i] = (unsigned int)l0 | ((unsigned int)l1 << 16);
        }
        const size_t m = (size_t)row0 + 16 * t + mHalf;        // multiple of 8
        *(v4u*)(BhT + (size_t)n * NN + m) = v4u{hw[0], hw[1], hw[2], hw[3]};
        *(v4u*)(BlT + (size_t)n * NN + m) = v4u{lw[0], lw[1], lw[2], lw[3]};
    }
}

// ---------------------------------------------------------------------------
// Stage 2: out = ELU( adj^2 @ support + bias ) via bf16x3 split-precision on
// v_wmma_f32_16x16x32_bf16:  a*b ~= ah*bh + al*bh + ah*bl  (f32 accumulate).
// HBM-bound: 268 MB read-once adj stream -> non-temporal b128 loads +
// global_prefetch 2 blocks ahead + double-buffered hi/lo LDS planes.
// ---------------------------------------------------------------------------
__global__ __launch_bounds__(512)
void gcn_aggregate_kernel(const float* __restrict__ ADJ,
                          const unsigned short* __restrict__ BhT,
                          const unsigned short* __restrict__ BlT,
                          const float* __restrict__ bias,
                          float* __restrict__ OUT) {
    __shared__ __align__(16) unsigned short ahT[2][MROWS * PADH];
    __shared__ __align__(16) unsigned short alT[2][MROWS * PADH];

    const int tid  = threadIdx.x;
    const int wave = tid >> 5, lane = tid & 31;
    const int laneM = lane & 15;
    const int kaOff = (lane >> 4) * 8;   // A: lanes>=16 start at K+8
    const int kbOff = (lane >> 4) * 8;   // B: lanes>=16 start at K+8
    const int row0  = blockIdx.x * MROWS;
    const int n0    = wave * 16;
    // stager: one b128 per thread per block: row rs, cols cs..cs+3
    const int rs = tid >> 4;
    const int cs = (tid & 15) * 4;
    const float* aBase  = ADJ + (size_t)(row0 + rs) * NN + cs;
    const int    ldsIdx = rs * PADH + cs;   // bf16 elements; byte addr 8-aligned

    // prologue: stage block 0 (square + hi/lo split fused into the stream)
    {
        v4f v = __builtin_nontemporal_load((const v4f*)aBase);
        v *= v;
        v2u h, l; split4(v, h, l);
        *(v2u*)&ahT[0][ldsIdx] = h;
        *(v2u*)&alT[0][ldsIdx] = l;
    }
    __syncthreads();

    v8f acc[MT] = {};
    constexpr int NB = NN / KBLK;  // 128
    const unsigned short* bhBase = BhT + (size_t)(n0 + laneM) * NN + kbOff;
    const unsigned short* blBase = BlT + (size_t)(n0 + laneM) * NN + kbOff;

    for (int kb = 0; kb < NB; ++kb) {
        const int  cur  = kb & 1;
        const bool more = (kb + 1 < NB);

        v2u sh, sl;
        if (more) {
            v4f v = __builtin_nontemporal_load(
                (const v4f*)(aBase + (size_t)(kb + 1) * KBLK));
            v *= v;
            split4(v, sh, sl);
        }
        if (kb + 2 < NB)   // keep the HBM stream 2 blocks deep
            __builtin_prefetch(aBase + (size_t)(kb + 2) * KBLK, 0, 1);

        const unsigned short* ahp = ahT[cur];
        const unsigned short* alp = alT[cur];

        #pragma unroll
        for (int k = 0; k < KBLK; k += 32) {
            const size_t kg = (size_t)kb * KBLK + k;
            // B operands (shared by both M-tiles): 2x b128 per plane
            v16bf Bh = make_bf16x16(*(const v4u*)(bhBase + kg),
                                    *(const v4u*)(bhBase + kg + 16));
            v16bf Bl = make_bf16x16(*(const v4u*)(blBase + kg),
                                    *(const v4u*)(blBase + kg + 16));
            #pragma unroll
            for (int t = 0; t < MT; ++t) {
                const unsigned short* ap = ahp + (16 * t + laneM) * PADH + k + kaOff;
                const unsigned short* lp = alp + (16 * t + laneM) * PADH + k + kaOff;
                v16bf Ah = make_bf16x16(*(const v4u*)ap, *(const v4u*)(ap + 16));
                v16bf Al = make_bf16x16(*(const v4u*)lp, *(const v4u*)(lp + 16));
                acc[t] = __builtin_amdgcn_wmma_f32_16x16x32_bf16(
                    false, Ah, false, Bh, (short)0, acc[t], false, false);
                acc[t] = __builtin_amdgcn_wmma_f32_16x16x32_bf16(
                    false, Al, false, Bh, (short)0, acc[t], false, false);
                acc[t] = __builtin_amdgcn_wmma_f32_16x16x32_bf16(
                    false, Ah, false, Bl, (short)0, acc[t], false, false);
            }
        }

        if (more) {
            *(v2u*)&ahT[cur ^ 1][ldsIdx] = sh;
            *(v2u*)&alT[cur ^ 1][ldsIdx] = sl;
        }
        __syncthreads();
    }

    // epilogue: bias + ELU, C/D layout VGPR r -> M = r + 8*(lane>=16), N = laneM
    const float bv    = bias[n0 + laneM];
    const int   mHalf = (lane >> 4) * 8;
    #pragma unroll
    for (int t = 0; t < MT; ++t) {
        #pragma unroll
        for (int r = 0; r < 8; ++r) {
            float v = acc[t][r] + bv;
            v = (v > 0.0f) ? v : expm1f(v);   // ELU(alpha=1)
            __builtin_nontemporal_store(
                v, &OUT[(size_t)(row0 + 16 * t + mHalf + r) * OUT_F + n0 + laneM]);
        }
    }
}

// ---------------------------------------------------------------------------
extern "C" void kernel_launch(void* const* d_in, const int* in_sizes, int n_in,
                              void* d_out, int out_size, void* d_ws, size_t ws_size,
                              hipStream_t stream) {
    const float* X    = (const float*)d_in[0];  // [8192, 512]
    const float* ADJ  = (const float*)d_in[1];  // [8192, 8192]
    const float* W    = (const float*)d_in[2];  // [512, 256]
    const float* bias = (const float*)d_in[3];  // [256]
    float*       OUT  = (float*)d_out;          // [8192, 256]

    // workspace: two transposed bf16 planes of support, 4 MB each
    unsigned short* BhT = (unsigned short*)d_ws;
    unsigned short* BlT = BhT + (size_t)OUT_F * NN;

    dim3 grid(NN / MROWS);   // 256 workgroups
    dim3 block(512);

    gcn_support_kernel  <<<grid, block, 0, stream>>>(X, W, BhT, BlT);
    gcn_aggregate_kernel<<<grid, block, 0, stream>>>(ADJ, BhT, BlT, bias, OUT);
}
